// StandGAT1_15839839387780
// MI455X (gfx1250) — matrix operand order repaired and checked by reading, not compiled, not run on hardware
//
#include <hip/hip_runtime.h>

typedef __attribute__((ext_vector_type(16))) __bf16 v16bf;
typedef __attribute__((ext_vector_type(8)))  float  v8f;

#define F_IN      512
#define F_OUT     256
#define NEG_SLOPE 0.2f
#define KT        (F_IN / 32)    // 16 K-tiles
#define NT        (F_OUT / 16)   // 16 N-tiles

#define WMMA_BF16(A, B, C) \
    __builtin_amdgcn_wmma_f32_16x16x32_bf16(false, (A), false, (B), (short)0, (C), false, false)

// ---------------------------------------------------------------------------
// Pre-split W[512,256] into packed bf16 hi/lo fragments laid out exactly as
// the per-lane WMMA B operand: wsf[(kt*NT+nt)*32 + lane][0:16]=hi, [16:32]=lo.
// B layout (16-bit, 32x16): elem e -> K = ((e>>3)<<4) + half*8 + (e&7); N = lane&15.
// 512 KB total -> L2 resident; GEMM loads it with plain b128s, no cvt.
// ---------------------------------------------------------------------------
__global__ __launch_bounds__(256) void gat_split_w(const float* __restrict__ W,
                                                   __bf16* __restrict__ wsf)
{
    const int t = blockIdx.x * blockDim.x + threadIdx.x;  // tile*32 + lane
    if (t >= KT * NT * 32) return;
    const int lane = t & 31;
    const int tile = t >> 5;            // kt*NT + nt
    const int nt   = tile & (NT - 1);
    const int kt   = tile >> 4;
    const int half = lane >> 4;
    const int n    = nt * 16 + (lane & 15);
    __bf16* dst = wsf + (size_t)t * 32;
#pragma unroll
    for (int e = 0; e < 16; ++e) {
        const int k = kt * 32 + ((e >> 3) << 4) + half * 8 + (e & 7);
        const float v  = W[(size_t)k * F_OUT + n];
        const __bf16 hi = (__bf16)v;
        dst[e]      = hi;
        dst[16 + e] = (__bf16)(v - (float)hi);
    }
}

// ---------------------------------------------------------------------------
// GEMM: h[N,256] = x[N,512] @ W via bf16x3 split WMMA.
// Grid (N/16), 128 threads = 4 waves. Block covers 16 rows x ALL 256 cols
// (x streamed from HBM exactly once). Each wave owns FOUR 16x16 output tiles:
// one shared A-fragment conversion feeds 12 WMMAs per wave per K-step.
// ---------------------------------------------------------------------------
__global__ __launch_bounds__(128) void gat_gemm_wmma(
    const float* __restrict__ x, const __bf16* __restrict__ wsf,
    float* __restrict__ h, int N)
{
    __shared__ __attribute__((aligned(32))) __bf16 aHi[32][16];
    __shared__ __attribute__((aligned(32))) __bf16 aLo[32][16];

    const int tid  = threadIdx.x;
    const int lane = tid & 31;               // wave32
    const int wave = tid >> 5;
    const int half = lane >> 4;
    const int m16  = lane & 15;

    const int M0  = blockIdx.x * 16;
    const int nt0 = wave * 4;                // this wave's four n-tiles

    // converter-thread mapping: this thread's float4 -> A-fragment slots.
    // A layout (16-bit, 16x32): lane = m + 16*half, elem e covers
    // K = ((e>>3)<<4) + half*8 + (e&7). Inverse for k0..k0+3 (same 8-run):
    const int arow  = tid >> 3;              // m: 0..15
    const int k0    = (tid & 7) * 4;         // 0,4,...,28
    const int chalf = (k0 >> 3) & 1;
    const int ce0   = (k0 & 7) + ((k0 >> 4) << 3);
    const int clane = arow + 16 * chalf;

    v8f acc0 = {}, acc1 = {}, acc2 = {}, acc3 = {};

    for (int kt = 0; kt < KT; ++kt) {
        // coalesced load + cooperative split of the 16x32 A tile (once/block)
        const float4 v = *(const float4*)(x + (size_t)(M0 + arow) * F_IN + kt * 32 + k0);
        const float f[4] = {v.x, v.y, v.z, v.w};
#pragma unroll
        for (int j = 0; j < 4; ++j) {
            const __bf16 hi = (__bf16)f[j];
            aHi[clane][ce0 + j] = hi;
            aLo[clane][ce0 + j] = (__bf16)(f[j] - (float)hi);
        }
        __syncthreads();

        const v16bf a_hi = *(const v16bf*)&aHi[lane][0];   // ds_load_b128 x2
        const v16bf a_lo = *(const v16bf*)&aLo[lane][0];

        const __bf16* bbase = wsf + ((size_t)(kt * NT + nt0) * 32 + lane) * 32;
        const v16bf b0_hi = *(const v16bf*)(bbase);
        const v16bf b0_lo = *(const v16bf*)(bbase + 16);
        const v16bf b1_hi = *(const v16bf*)(bbase + 1024);      // next tile: 32 lanes*32
        const v16bf b1_lo = *(const v16bf*)(bbase + 1024 + 16);
        const v16bf b2_hi = *(const v16bf*)(bbase + 2048);
        const v16bf b2_lo = *(const v16bf*)(bbase + 2048 + 16);
        const v16bf b3_hi = *(const v16bf*)(bbase + 3072);
        const v16bf b3_lo = *(const v16bf*)(bbase + 3072 + 16);

        // bf16x3: hi*hi + hi*lo + lo*hi for each of the four N tiles
        acc0 = WMMA_BF16(a_hi, b0_hi, acc0);
        acc0 = WMMA_BF16(a_hi, b0_lo, acc0);
        acc0 = WMMA_BF16(a_lo, b0_hi, acc0);
        acc1 = WMMA_BF16(a_hi, b1_hi, acc1);
        acc1 = WMMA_BF16(a_hi, b1_lo, acc1);
        acc1 = WMMA_BF16(a_lo, b1_hi, acc1);
        acc2 = WMMA_BF16(a_hi, b2_hi, acc2);
        acc2 = WMMA_BF16(a_hi, b2_lo, acc2);
        acc2 = WMMA_BF16(a_lo, b2_hi, acc2);
        acc3 = WMMA_BF16(a_hi, b3_hi, acc3);
        acc3 = WMMA_BF16(a_hi, b3_lo, acc3);
        acc3 = WMMA_BF16(a_lo, b3_hi, acc3);

        __syncthreads();
    }

    // C/D layout: VGPR r -> M = r + 8*half; N = lane&15
    const int c0 = (nt0 + 0) * 16 + m16;
    const int c1 = (nt0 + 1) * 16 + m16;
    const int c2 = (nt0 + 2) * 16 + m16;
    const int c3 = (nt0 + 3) * 16 + m16;
#pragma unroll
    for (int r = 0; r < 8; ++r) {
        const size_t row = (size_t)(M0 + r + half * 8) * F_OUT;
        h[row + c0] = acc0[r];
        h[row + c1] = acc1[r];
        h[row + c2] = acc2[r];
        h[row + c3] = acc3[r];
    }
}

// ---------------------------------------------------------------------------
// Per-node attention terms: asrc[n] = h[n,:]·att_src, adst[n] = h[n,:]·att_dst
// One wave per node, 8 features per lane, shuffle reduction.
// ---------------------------------------------------------------------------
__global__ __launch_bounds__(256) void gat_att(
    const float* __restrict__ h, const float* __restrict__ att_src,
    const float* __restrict__ att_dst, float* __restrict__ asrc,
    float* __restrict__ adst, int N)
{
    const int lane = threadIdx.x & 31;
    const int node = blockIdx.x * 8 + (threadIdx.x >> 5);
    if (node >= N) return;
    float ss = 0.f, sd = 0.f;
#pragma unroll
    for (int j = 0; j < 8; ++j) {
        const int f = lane + 32 * j;
        const float hv = h[(size_t)node * F_OUT + f];
        ss += hv * att_src[f];
        sd += hv * att_dst[f];
    }
#pragma unroll
    for (int off = 16; off > 0; off >>= 1) {
        ss += __shfl_xor(ss, off, 32);
        sd += __shfl_xor(sd, off, 32);
    }
    if (lane == 0) { asrc[node] = ss; adst[node] = sd; }
}

// ---------------------------------------------------------------------------
// Init: out = 0, m = -inf, denom = 0 (harness poisons buffers; re-init per call)
// ---------------------------------------------------------------------------
__global__ void gat_init(float* __restrict__ out, float* __restrict__ m,
                         float* __restrict__ denom, int N)
{
    const int stride = gridDim.x * blockDim.x;
    const int i0 = blockIdx.x * blockDim.x + threadIdx.x;
    const int total = N * F_OUT;
    for (int i = i0; i < total; i += stride) out[i] = 0.f;
    for (int i = i0; i < N; i += stride) {
        m[i]     = -__builtin_huge_valf();
        denom[i] = 0.f;
    }
}

__device__ __forceinline__ void atomicMaxF32(float* addr, float val) {
    if (val >= 0.f) atomicMax((int*)addr, __float_as_int(val));
    else            atomicMin((unsigned int*)addr, __float_as_uint(val));
}

// ---------------------------------------------------------------------------
// Edge logits + leaky ReLU + per-destination running max. Edges [E,E+N) are
// the implicit self-loops.
// ---------------------------------------------------------------------------
__global__ void gat_edge_logits(const long long* __restrict__ adj, int E, int N,
                                const float* __restrict__ asrc,
                                const float* __restrict__ adst,
                                float* __restrict__ e, float* __restrict__ m)
{
    const int total = E + N;
    const int stride = gridDim.x * blockDim.x;
    for (int i = blockIdx.x * blockDim.x + threadIdx.x; i < total; i += stride) {
        int s, d;
        if (i < E) { s = (int)adj[i]; d = (int)adj[E + i]; }
        else       { s = d = i - E; }
        float ev = asrc[s] + adst[d];
        ev = ev > 0.f ? ev : NEG_SLOPE * ev;
        e[i] = ev;
        atomicMaxF32(&m[d], ev);
    }
}

// ---------------------------------------------------------------------------
// ex = exp(e - m[dst]); denom[dst] += ex   (ex overwrites e)
// ---------------------------------------------------------------------------
__global__ void gat_edge_exp(const long long* __restrict__ adj, int E, int N,
                             const float* __restrict__ m,
                             float* __restrict__ e, float* __restrict__ denom)
{
    const int total = E + N;
    const int stride = gridDim.x * blockDim.x;
    for (int i = blockIdx.x * blockDim.x + threadIdx.x; i < total; i += stride) {
        const int d = (i < E) ? (int)adj[E + i] : (i - E);
        const float ex = __expf(e[i] - m[d]);
        e[i] = ex;
        atomicAdd(&denom[d], ex);
    }
}

// ---------------------------------------------------------------------------
// out[dst,:] += (ex/denom[dst]) * h[src,:]  — one block (=F_OUT lanes) per edge
// ---------------------------------------------------------------------------
__global__ __launch_bounds__(F_OUT) void gat_scatter(
    const long long* __restrict__ adj, int E, int N,
    const float* __restrict__ h, const float* __restrict__ e,
    const float* __restrict__ denom, float* __restrict__ out)
{
    const int i = blockIdx.x;
    int s, d;
    if (i < E) { s = (int)adj[i]; d = (int)adj[E + i]; }
    else       { s = d = i - E; }
    const float alpha = e[i] / denom[d];
    const int t = threadIdx.x;
    atomicAdd(&out[(size_t)d * F_OUT + t], alpha * h[(size_t)s * F_OUT + t]);
}

// ---------------------------------------------------------------------------
// out = relu(out + bias)
// ---------------------------------------------------------------------------
__global__ void gat_finalize(float* __restrict__ out,
                             const float* __restrict__ bias, int N)
{
    const int total = N * F_OUT;
    const int stride = gridDim.x * blockDim.x;
    for (int i = blockIdx.x * blockDim.x + threadIdx.x; i < total; i += stride) {
        const float v = out[i] + bias[i & (F_OUT - 1)];
        out[i] = v > 0.f ? v : 0.f;
    }
}

extern "C" void kernel_launch(void* const* d_in, const int* in_sizes, int n_in,
                              void* d_out, int out_size, void* d_ws, size_t ws_size,
                              hipStream_t stream)
{
    const float*     x       = (const float*)d_in[0];
    const long long* adj     = (const long long*)d_in[1];   // [2,E] int64
    const float*     W       = (const float*)d_in[2];
    const float*     att_src = (const float*)d_in[3];
    const float*     att_dst = (const float*)d_in[4];
    const float*     bias    = (const float*)d_in[5];

    const int N    = in_sizes[0] / F_IN;   // 50000 (== 3125*16)
    const int E    = in_sizes[1] / 2;      // 400000
    const int Etot = E + N;

    // workspace layout (floats): h | asrc | adst | e | m | denom | wsf(bf16)
    float* ws    = (float*)d_ws;
    float* h     = ws;
    float* asrc  = h + (size_t)N * F_OUT;
    float* adst  = asrc + N;
    float* e     = adst + N;
    float* m     = e + Etot;
    float* denom = m + N;
    size_t fcnt  = (size_t)(denom + N - ws);
    fcnt = (fcnt + 15) & ~(size_t)15;                 // 64B-align fragment region
    __bf16* wsf  = (__bf16*)(ws + fcnt);              // KT*NT*32*32 bf16 = 512 KB
    float* out   = (float*)d_out;

    gat_split_w<<<(KT * NT * 32 + 255) / 256, 256, 0, stream>>>(W, wsf);

    gat_gemm_wmma<<<N / 16, 128, 0, stream>>>(x, wsf, h, N);

    gat_att<<<(N + 7) / 8, 256, 0, stream>>>(h, att_src, att_dst, asrc, adst, N);
    gat_init<<<1024, 256, 0, stream>>>(out, m, denom, N);
    gat_edge_logits<<<(Etot + 255) / 256, 256, 0, stream>>>(adj, E, N, asrc, adst, e, m);
    gat_edge_exp<<<(Etot + 255) / 256, 256, 0, stream>>>(adj, E, N, m, e, denom);
    gat_scatter<<<Etot, F_OUT, 0, stream>>>(adj, E, N, h, e, denom, out);
    gat_finalize<<<2048, 256, 0, stream>>>(out, bias, N);
}
